// GroupExpertChoiceMoELayer_35725537968820
// MI455X (gfx1250) — matrix operand order repaired and checked by reading, not compiled, not run on hardware
//
#include <hip/hip_runtime.h>

#define N_TOK 8192
#define H_DIM 1024
#define E_NUM 8
#define KSEL  1024
#define F_DIM 2730
#define FPAD  2816   // 22 * 128, multiple of 32 for the K loop of the down GEMM
#define LSTR  40     // LDS tile row stride in bf16 elements (80B, 16B aligned rows)

typedef __attribute__((ext_vector_type(16))) __bf16 v16bf;
typedef __attribute__((ext_vector_type(8)))  float  v8f;

__device__ __forceinline__ unsigned short f2bf(float f) {
    unsigned int u = __float_as_uint(f);
    u += 0x7FFFu + ((u >> 16) & 1u);          // round-to-nearest-even
    return (unsigned short)(u >> 16);
}

// A fragment (16x32 bf16, ISA 7.12.2 "16-bit A-Matrix 16x32"):
// lanes 0-15 hold K {0..7, 16..23}, lanes 16-31 hold K {8..15, 24..31}, pairs per VGPR.
// From LDS row (row = M, stride LSTR): 16B at byte 16*half and 16B at byte 32+16*half.
__device__ __forceinline__ v16bf ldsFragA(const unsigned short* rowBase, int hoff16) {
    union { v16bf v; uint4 q[2]; } f;
    const char* p = (const char*)rowBase + hoff16;
    f.q[0] = *(const uint4*)(p);
    f.q[1] = *(const uint4*)(p + 32);
    return f.v;
}

// B fragment (32x16 bf16, per ISA sparse-B pattern 7.12.4/7.12.5):
// lanes 0-15 hold K {0..7, 16..23}, lanes 16-31 hold K {8..15, 24..31}, pairs per VGPR.
// From LDS B^T row (row = N, stride LSTR): 16B at byte 16*half and 16B at byte 32+16*half.
__device__ __forceinline__ v16bf ldsFragB(const unsigned short* rowBase, int boff32) {
    union { v16bf v; uint4 q[2]; } f;
    const char* p = (const char*)rowBase + boff32;
    f.q[0] = *(const uint4*)(p);
    f.q[1] = *(const uint4*)(p + 32);
    return f.v;
}

// ---------------- Router: logits = x @ Wr + br ; softmax -> Sm (N_TOK x 8) ----------------
__global__ __launch_bounds__(256) void router_kernel(const float* __restrict__ x,
                                                     const float* __restrict__ Wr,
                                                     const float* __restrict__ br,
                                                     float* __restrict__ Sm) {
    __shared__ float sWr[H_DIM * E_NUM];
    const int tid = threadIdx.x;
    for (int i = tid; i < H_DIM * E_NUM; i += 256) sWr[i] = Wr[i];
    __syncthreads();
    const int lane = tid & 31;
    const int tok  = blockIdx.x * 8 + (tid >> 5);
    const float* xr = x + (size_t)tok * H_DIM;
    float p[E_NUM];
#pragma unroll
    for (int e = 0; e < E_NUM; ++e) p[e] = 0.f;
    for (int h = lane; h < H_DIM; h += 32) {
        const float xv = xr[h];
        const float* w = &sWr[h * E_NUM];
#pragma unroll
        for (int e = 0; e < E_NUM; ++e) p[e] += xv * w[e];
    }
#pragma unroll
    for (int s = 16; s > 0; s >>= 1) {
#pragma unroll
        for (int e = 0; e < E_NUM; ++e) p[e] += __shfl_xor(p[e], s, 32);
    }
    if (lane == 0) {
        float mx = -1e30f;
#pragma unroll
        for (int e = 0; e < E_NUM; ++e) { p[e] += br[e]; mx = fmaxf(mx, p[e]); }
        float ex[E_NUM]; float sum = 0.f;
#pragma unroll
        for (int e = 0; e < E_NUM; ++e) { ex[e] = __expf(p[e] - mx); sum += ex[e]; }
        const float inv = 1.0f / sum;
#pragma unroll
        for (int e = 0; e < E_NUM; ++e) Sm[(size_t)tok * E_NUM + e] = ex[e] * inv;
    }
}

// ---- Exact expert-choice top-k per expert: radix-select threshold + deterministic compaction ----
__global__ __launch_bounds__(256) void topk_kernel(const float* __restrict__ Sm,
                                                   int* __restrict__ idxb,
                                                   float* __restrict__ Gb) {
    const int e   = blockIdx.x;
    const int tid = threadIdx.x;
    __shared__ unsigned int hist[256];
    __shared__ unsigned int sh_pref;
    __shared__ int sh_r;
    if (tid == 0) { sh_pref = 0u; sh_r = KSEL; }
    __syncthreads();
    for (int bshift = 3; bshift >= 0; --bshift) {
        hist[tid] = 0u;
        __syncthreads();
        const unsigned int pref = sh_pref;
        for (int t = tid; t < N_TOK; t += 256) {
            const unsigned int key = __float_as_uint(Sm[(size_t)t * E_NUM + e]);
            if (bshift == 3 || (key >> ((bshift + 1) * 8)) == pref)
                atomicAdd(&hist[(key >> (bshift * 8)) & 255u], 1u);
        }
        __syncthreads();
        if (tid == 0) {
            const int r = sh_r;
            unsigned int cum = 0; int c = 255;
            for (; c > 0; --c) {
                if (cum + hist[c] >= (unsigned int)r) break;
                cum += hist[c];
            }
            sh_r    = r - (int)cum;
            sh_pref = (sh_pref << 8) | (unsigned int)c;
        }
        __syncthreads();
    }
    if (tid < 32) {   // single wave: deterministic, index-ordered tie selection
        const unsigned int thresh = sh_pref;
        const int quota = sh_r;
        const int lane  = tid;
        const unsigned int lmask = (1u << lane) - 1u;
        int selBase = 0, eqBase = 0;
        for (int base = 0; base < N_TOK; base += 32) {
            const int t = base + lane;
            const float v = Sm[(size_t)t * E_NUM + e];
            const unsigned int key = __float_as_uint(v);
            const bool gt = key > thresh;
            const bool eq = (key == thresh);
            const unsigned int eqb = __builtin_amdgcn_ballot_w32(eq);
            const int eqRank = eqBase + __builtin_popcount(eqb & lmask);
            const bool sel = gt || (eq && eqRank < quota);
            const unsigned int selb = __builtin_amdgcn_ballot_w32(sel);
            if (sel) {
                const int pos = selBase + __builtin_popcount(selb & lmask);
                idxb[e * KSEL + pos] = t;
                Gb[e * KSEL + pos]   = v;
            }
            selBase += __builtin_popcount(selb);
            eqBase  += __builtin_popcount(eqb);
        }
    }
}

// ---------------- slot table for deterministic (atomic-free) combine ----------------
__global__ void slot_init_kernel(int* __restrict__ slotTab) {
    const int i = blockIdx.x * 256 + threadIdx.x;
    if (i < N_TOK * E_NUM) slotTab[i] = -1;
}
__global__ void slot_fill_kernel(const int* __restrict__ idxb, int* __restrict__ slotTab) {
    const int i = blockIdx.x * 256 + threadIdx.x;
    if (i < E_NUM * KSEL) {
        const int t = idxb[i];
        slotTab[t * E_NUM + (i >> 10)] = i & 1023;   // (token, expert) pairs are unique
    }
}

// ---------------- Up-proj + SiLU: mid[e] = silu(gather(x) @ Wup[g] + bup[g]) (bf16) ----------------
__global__ __launch_bounds__(256) void moe_up_kernel(const float* __restrict__ x,
                                                     const int* __restrict__ idxb,
                                                     const float* __restrict__ Wup,
                                                     const float* __restrict__ bup,
                                                     unsigned short* __restrict__ mid) {
    __shared__ __align__(16) unsigned short As[2][128 * LSTR];
    __shared__ __align__(16) unsigned short Bs[2][128 * LSTR];
    const int tid   = threadIdx.x;
    const int e     = blockIdx.z;
    const int g     = e >> 1;
    const int n0    = blockIdx.x * 128;
    const int m0    = blockIdx.y * 128;
    const int lane  = tid & 31, wave = tid >> 5;
    const int waveM = wave >> 1, waveN = wave & 1;
    const int kq4 = (tid & 7) * 4,  r0 = tid >> 3;
    const int nq4 = (tid & 31) * 4, kr = tid >> 5;
    const bool fullN = (n0 + 128 <= F_DIM);
    const size_t wupBase = (size_t)g * H_DIM * F_DIM;
    int tok[4];
#pragma unroll
    for (int p = 0; p < 4; ++p) tok[p] = idxb[e * KSEL + m0 + r0 + 32 * p];

    auto loadT = [&](int k0, int buf) {
#pragma unroll
        for (int p = 0; p < 4; ++p) {             // A: gathered token rows, fp32 -> bf16, packed
            const float4 v = *(const float4*)(x + (size_t)tok[p] * H_DIM + k0 + kq4);
            uint2 pk;
            pk.x = (unsigned)f2bf(v.x) | ((unsigned)f2bf(v.y) << 16);
            pk.y = (unsigned)f2bf(v.z) | ((unsigned)f2bf(v.w) << 16);
            *(uint2*)&As[buf][(r0 + 32 * p) * LSTR + kq4] = pk;
        }
#pragma unroll
        for (int p = 0; p < 4; ++p) {             // B: Wup rows, stored transposed [n][k]
            const int k = kr + 8 * p;
            const float* src = Wup + wupBase + (size_t)(k0 + k) * F_DIM + (n0 + nq4);
            float vx, vy, vz, vw;
            if (fullN) {
                const float2 a = *(const float2*)src;       // F_DIM even -> 8B aligned
                const float2 b = *(const float2*)(src + 2);
                vx = a.x; vy = a.y; vz = b.x; vw = b.y;
            } else {
                vx = (n0 + nq4 + 0 < F_DIM) ? src[0] : 0.f;
                vy = (n0 + nq4 + 1 < F_DIM) ? src[1] : 0.f;
                vz = (n0 + nq4 + 2 < F_DIM) ? src[2] : 0.f;
                vw = (n0 + nq4 + 3 < F_DIM) ? src[3] : 0.f;
            }
            Bs[buf][(nq4 + 0) * LSTR + k] = f2bf(vx);
            Bs[buf][(nq4 + 1) * LSTR + k] = f2bf(vy);
            Bs[buf][(nq4 + 2) * LSTR + k] = f2bf(vz);
            Bs[buf][(nq4 + 3) * LSTR + k] = f2bf(vw);
        }
    };

    v8f acc[2][4];
    const v8f zero8 = {0.f, 0.f, 0.f, 0.f, 0.f, 0.f, 0.f, 0.f};
#pragma unroll
    for (int im = 0; im < 2; ++im)
#pragma unroll
        for (int in = 0; in < 4; ++in) acc[im][in] = zero8;

    const int mloc = lane & 15;
    const int hoff16 = (lane >> 4) * 16;
    const int boff32 = (lane >> 4) * 16;
    const int KT = H_DIM / 32;
    for (int kt = 0; kt < KT; ++kt) {
        const int buf = kt & 1;
        if (kt == 0) { loadT(0, 0); __syncthreads(); }
        if (kt + 1 < KT) loadT((kt + 1) * 32, buf ^ 1);
        v16bf aF[2], bF[4];
#pragma unroll
        for (int im = 0; im < 2; ++im)
            aF[im] = ldsFragA(&As[buf][(waveM * 32 + im * 16 + mloc) * LSTR], hoff16);
#pragma unroll
        for (int in = 0; in < 4; ++in)
            bF[in] = ldsFragB(&Bs[buf][(waveN * 64 + in * 16 + mloc) * LSTR], boff32);
#pragma unroll
        for (int im = 0; im < 2; ++im)
#pragma unroll
            for (int in = 0; in < 4; ++in)
                acc[im][in] = __builtin_amdgcn_wmma_f32_16x16x32_bf16(
                    false, aF[im], false, bF[in], (short)0, acc[im][in], false, false);
        __syncthreads();
    }

    const int half = lane >> 4;
#pragma unroll
    for (int im = 0; im < 2; ++im) {
        const int mBase = m0 + waveM * 32 + im * 16 + half * 8;
#pragma unroll
        for (int in = 0; in < 4; ++in) {
            const int nG = n0 + waveN * 64 + in * 16 + mloc;
            const float bias = (nG < F_DIM) ? bup[g * F_DIM + nG] : 0.f;
#pragma unroll
            for (int r = 0; r < 8; ++r) {
                const float v = acc[im][in][r] + bias;
                const float s = v / (1.0f + __expf(-v));     // SiLU; pad region -> silu(0)=0
                mid[((size_t)e * KSEL + mBase + r) * FPAD + nG] = f2bf(s);
            }
        }
    }
}

// ------------- Down-proj: out[e] = (mid[e] @ Wdown[e] + bdown[e]) * G[e] (fp32) -------------
__global__ __launch_bounds__(256) void moe_down_kernel(const unsigned short* __restrict__ mid,
                                                       const float* __restrict__ Wdown,
                                                       const float* __restrict__ bdown,
                                                       const float* __restrict__ Gb,
                                                       float* __restrict__ outbuf) {
    __shared__ __align__(16) unsigned short As[2][128 * LSTR];
    __shared__ __align__(16) unsigned short Bs[2][128 * LSTR];
    const int tid   = threadIdx.x;
    const int e     = blockIdx.z;
    const int n0    = blockIdx.x * 128;
    const int m0    = blockIdx.y * 128;
    const int lane  = tid & 31, wave = tid >> 5;
    const int waveM = wave >> 1, waveN = wave & 1;
    const int kq4 = (tid & 7) * 4,  r0 = tid >> 3;
    const int nq4 = (tid & 31) * 4, kr = tid >> 5;
    const size_t midBase = (size_t)e * KSEL;
    const size_t wdBase  = (size_t)e * F_DIM * H_DIM;

    auto loadT = [&](int k0, int buf) {
#pragma unroll
        for (int p = 0; p < 4; ++p) {   // A: mid rows already bf16 -> async global->LDS (ASYNCcnt)
            const unsigned short* gsrc =
                mid + (midBase + m0 + r0 + 32 * p) * FPAD + k0 + kq4;
            const unsigned int ldsAddr =
                (unsigned int)(size_t)&As[buf][(r0 + 32 * p) * LSTR + kq4];
            asm volatile("global_load_async_to_lds_b64 %0, %1, off"
                         :: "v"(ldsAddr), "v"(gsrc) : "memory");
        }
#pragma unroll
        for (int p = 0; p < 4; ++p) {             // B: Wdown rows (k < F_DIM), transposed [n][k]
            const int k  = kr + 8 * p;
            const int kk = k0 + k;
            float4 v = {0.f, 0.f, 0.f, 0.f};
            if (kk < F_DIM) v = *(const float4*)(Wdown + wdBase + (size_t)kk * H_DIM + n0 + nq4);
            Bs[buf][(nq4 + 0) * LSTR + k] = f2bf(v.x);
            Bs[buf][(nq4 + 1) * LSTR + k] = f2bf(v.y);
            Bs[buf][(nq4 + 2) * LSTR + k] = f2bf(v.z);
            Bs[buf][(nq4 + 3) * LSTR + k] = f2bf(v.w);
        }
    };

    v8f acc[2][4];
    const v8f zero8 = {0.f, 0.f, 0.f, 0.f, 0.f, 0.f, 0.f, 0.f};
#pragma unroll
    for (int im = 0; im < 2; ++im)
#pragma unroll
        for (int in = 0; in < 4; ++in) acc[im][in] = zero8;

    const int mloc = lane & 15;
    const int hoff16 = (lane >> 4) * 16;
    const int boff32 = (lane >> 4) * 16;
    const int KT = FPAD / 32;   // 88
    for (int kt = 0; kt < KT; ++kt) {
        const int buf = kt & 1;
        if (kt == 0) {
            loadT(0, 0);
            asm volatile("s_wait_asynccnt 0x0" ::: "memory");
            __syncthreads();
        }
        if (kt + 1 < KT) loadT((kt + 1) * 32, buf ^ 1);
        v16bf aF[2], bF[4];
#pragma unroll
        for (int im = 0; im < 2; ++im)
            aF[im] = ldsFragA(&As[buf][(waveM * 32 + im * 16 + mloc) * LSTR], hoff16);
#pragma unroll
        for (int in = 0; in < 4; ++in)
            bF[in] = ldsFragB(&Bs[buf][(waveN * 64 + in * 16 + mloc) * LSTR], boff32);
#pragma unroll
        for (int im = 0; im < 2; ++im)
#pragma unroll
            for (int in = 0; in < 4; ++in)
                acc[im][in] = __builtin_amdgcn_wmma_f32_16x16x32_bf16(
                    false, aF[im], false, bF[in], (short)0, acc[im][in], false, false);
        asm volatile("s_wait_asynccnt 0x0" ::: "memory");   // next-buffer async A complete
        __syncthreads();
    }

    const int half = lane >> 4;
#pragma unroll
    for (int im = 0; im < 2; ++im) {
        const int mBase = m0 + waveM * 32 + im * 16 + half * 8;
        float gs[8];
#pragma unroll
        for (int r = 0; r < 8; ++r) gs[r] = Gb[e * KSEL + mBase + r];
#pragma unroll
        for (int in = 0; in < 4; ++in) {
            const int nG = n0 + waveN * 64 + in * 16 + mloc;
            const float bd = bdown[e * H_DIM + nG];
#pragma unroll
            for (int r = 0; r < 8; ++r)
                outbuf[((size_t)e * KSEL + mBase + r) * H_DIM + nG] =
                    (acc[im][in][r] + bd) * gs[r];
        }
    }
}

// ------------- Deterministic gated combine: y[t] = sum_e out[e][slot(t,e)] -------------
__global__ __launch_bounds__(256) void combine_kernel(const float* __restrict__ outbuf,
                                                      const int* __restrict__ slotTab,
                                                      float* __restrict__ y) {
    const int t   = blockIdx.x;
    const int tid = threadIdx.x;
    __shared__ int slots[E_NUM];
    if (tid < E_NUM) slots[tid] = slotTab[t * E_NUM + tid];
    __syncthreads();
    const int h4 = tid * 4;
    float4 a = {0.f, 0.f, 0.f, 0.f};
#pragma unroll
    for (int e = 0; e < E_NUM; ++e) {
        const int s = slots[e];
        if (s >= 0) {
            const float4 v = *(const float4*)(outbuf + ((size_t)e * KSEL + s) * H_DIM + h4);
            a.x += v.x; a.y += v.y; a.z += v.z; a.w += v.w;
        }
    }
    *(float4*)(y + (size_t)t * H_DIM + h4) = a;
}

extern "C" void kernel_launch(void* const* d_in, const int* in_sizes, int n_in,
                              void* d_out, int out_size, void* d_ws, size_t ws_size,
                              hipStream_t stream) {
    const float* x     = (const float*)d_in[0];
    const float* Wr    = (const float*)d_in[1];
    const float* br    = (const float*)d_in[2];
    const float* Wup   = (const float*)d_in[3];
    const float* bup   = (const float*)d_in[4];
    const float* Wdown = (const float*)d_in[5];
    const float* bdown = (const float*)d_in[6];
    float* y = (float*)d_out;

    char* ws = (char*)d_ws;
    float*          Sm      = (float*)(ws + 0);                 //   256 KB
    int*            idxb    = (int*)(ws + 262144);              //    32 KB
    float*          Gb      = (float*)(ws + 294912);            //    32 KB
    int*            slotTab = (int*)(ws + 327680);              //   256 KB
    unsigned short* mid     = (unsigned short*)(ws + 589824);   //  ~44 MB  (8*1024*2816 bf16)
    float*          outbuf  = (float*)(ws + 589824 + 46137344); //  ~32 MB  (8*1024*1024 f32)

    router_kernel<<<N_TOK / 8, 256, 0, stream>>>(x, Wr, br, Sm);
    topk_kernel<<<E_NUM, 256, 0, stream>>>(Sm, idxb, Gb);
    slot_init_kernel<<<(N_TOK * E_NUM + 255) / 256, 256, 0, stream>>>(slotTab);
    slot_fill_kernel<<<(E_NUM * KSEL + 255) / 256, 256, 0, stream>>>(idxb, slotTab);
    moe_up_kernel<<<dim3(FPAD / 128, KSEL / 128, E_NUM), 256, 0, stream>>>(x, idxb, Wup, bup, mid);
    moe_down_kernel<<<dim3(H_DIM / 128, KSEL / 128, E_NUM), 256, 0, stream>>>(mid, Wdown, bdown, Gb, outbuf);
    combine_kernel<<<N_TOK, 256, 0, stream>>>(outbuf, slotTab, y);
}
